// SingleSampleLSTMLayer_609885356757
// MI455X (gfx1250) — matrix operand order repaired and checked by reading, not compiled, not run on hardware
//
#include <hip/hip_runtime.h>
#include <hip/hip_bf16.h>

// Problem constants (match reference)
#define Bm 32
#define Nn 256
#define Tt 1024
#define Hh 64
#define Gg 256   // 4*H
#define HP 80    // padded row length in halfs (160B, 16B-aligned rows)

typedef _Float16 v16h __attribute__((ext_vector_type(16)));
typedef _Float16 v8h  __attribute__((ext_vector_type(8)));
typedef float    v8f  __attribute__((ext_vector_type(8)));
typedef float    v4f  __attribute__((ext_vector_type(4)));

#define LOG2E 1.4426950408889634f

// ---- fast transcendental helpers (hardware TRANS ops) ----
__device__ __forceinline__ float fast_exp2(float x) {
#if __has_builtin(__builtin_amdgcn_exp2f)
    return __builtin_amdgcn_exp2f(x);      // v_exp_f32
#else
    return __expf(x * 0.6931471805599453f);
#endif
}
__device__ __forceinline__ float fast_rcp(float x) {
#if __has_builtin(__builtin_amdgcn_rcpf)
    return __builtin_amdgcn_rcpf(x);       // v_rcp_f32
#else
    return 1.0f / x;
#endif
}
// inputs already pre-scaled by LOG2E (sigmoid) / 2*LOG2E (tanh) in the weights
__device__ __forceinline__ float sig_pre(float xs)  { return fast_rcp(1.0f + fast_exp2(-xs)); }
__device__ __forceinline__ float tanh_pre(float xs2){ return 2.0f * fast_rcp(1.0f + fast_exp2(-xs2)) - 1.0f; }
// natural-units tanh for tanh(c_new)
__device__ __forceinline__ float tanh_nat(float x) {
#if __has_builtin(__builtin_amdgcn_tanhf)
    return __builtin_amdgcn_tanhf(x);      // v_tanh_f32 (gfx1250)
#else
    return 2.0f * fast_rcp(1.0f + fast_exp2(-2.0f * LOG2E * x)) - 1.0f;
#endif
}

// ---- CDNA5 split workgroup barrier (s_barrier_signal / s_barrier_wait) ----
__device__ __forceinline__ void wg_barrier_signal() {
    // drain our LDS stores so they are visible, then arrive
    asm volatile("s_wait_dscnt 0x0\n\ts_barrier_signal -1" ::: "memory");
}
__device__ __forceinline__ void wg_barrier_wait() {
    asm volatile("s_barrier_wait -1" ::: "memory");
}

// B-operand fragment for V_WMMA_F32_16X16X32_F16 (B is 32K x 16N):
// lane holds 16 consecutive K values (f16) of one column g.  Scaled on convert.
__device__ __forceinline__ v16h load_wfrag_g(const float* __restrict__ Wn, int g, int kb, float scale) {
    const float* p = Wn + g * Hh + kb;
    v4f f0 = *(const v4f*)(p + 0);
    v4f f1 = *(const v4f*)(p + 4);
    v4f f2 = *(const v4f*)(p + 8);
    v4f f3 = *(const v4f*)(p + 12);
    v16h r;
#pragma unroll
    for (int i = 0; i < 4; ++i) {
        r[i]      = (_Float16)(f0[i] * scale);
        r[4 + i]  = (_Float16)(f1[i] * scale);
        r[8 + i]  = (_Float16)(f2[i] * scale);
        r[12 + i] = (_Float16)(f3[i] * scale);
    }
    return r;
}

// Same B fragment but from LDS-resident f16 weights (padded rows of HP halfs).
__device__ __forceinline__ v16h load_bfrag_lds(const _Float16* wbuf, int g, int kb) {
    const _Float16* p = wbuf + g * HP + kb;
    v8h lo = *(const v8h*)(p);
    v8h hi = *(const v8h*)(p + 8);        // consecutive 16 K values
    return __builtin_shufflevector(lo, hi, 0, 1, 2, 3, 4, 5, 6, 7,
                                           8, 9, 10, 11, 12, 13, 14, 15);
}

// A-operand fragment (16M x 32K f16): lane = row m; VGPR0-7 hold K=kbase..+15
// (kbase already includes the lane-half offset).
__device__ __forceinline__ v16h load_afrag(const _Float16* hbuf, int m, int kbase) {
    const _Float16* p = hbuf + m * HP + kbase;
    v8h lo = *(const v8h*)(p);
    v8h hi = *(const v8h*)(p + 8);
    return __builtin_shufflevector(lo, hi, 0, 1, 2, 3, 4, 5, 6, 7,
                                           8, 9, 10, 11, 12, 13, 14, 15);
}

// One full LSTM timestep (both layers).  h0r/h0w/h1r/h1w are compile-time
// constant LDS addresses at each call site (phase hardcoded by the caller),
// so all DS accesses lower to base + immediate offset with no per-step
// address arithmetic.
__device__ __forceinline__ void lstm_step(
    const _Float16* __restrict__ h0r, _Float16* __restrict__ h0w,
    const _Float16* __restrict__ h1r, _Float16* __restrict__ h1w,
    const _Float16* __restrict__ sw1p, const _Float16* __restrict__ sw2p,
    const float (&sxTt)[Bm], float wfct,
    const v16h (&wb0)[4][2], const float (&bias0)[4], const float (&bias1)[4],
    const float (&wxc)[4],
    float (&c0)[8], float (&c1)[8], float (&acc)[8],
    int arow, int mrow, int u, int lh)
{
    // A fragments of previous-step h0 / h1 (no WAR barrier needed: double buffer)
    v16h a0[2], a1[2];
#pragma unroll
    for (int ks = 0; ks < 2; ++ks) {
        a0[ks] = load_afrag(h0r, arow, ks * 32 + lh * 16);
        a1[ks] = load_afrag(h1r, arow, ks * 32 + lh * 16);
    }

    float xv[8];
#pragma unroll
    for (int v = 0; v < 8; ++v) xv[v] = sxTt[mrow + v];   // LDS broadcast

    // ---------- layer 0: gates = x*wx0 + h0 @ Whh0^T + b0 (exp2 domain) ----------
    v8f gate[4];
#pragma unroll
    for (int q = 0; q < 4; ++q) {
        v8f c;
#pragma unroll
        for (int v = 0; v < 8; ++v) c[v] = bias0[q] + xv[v] * wxc[q];
        c = __builtin_amdgcn_wmma_f32_16x16x32_f16(false, a0[0], false, wb0[q][0], (short)0, c, false, false);
        c = __builtin_amdgcn_wmma_f32_16x16x32_f16(false, a0[1], false, wb0[q][1], (short)0, c, false, false);
        gate[q] = c;
    }
#pragma unroll
    for (int v = 0; v < 8; ++v) {
        float ig = sig_pre(gate[0][v]);
        float fg = sig_pre(gate[1][v]);
        float gg = tanh_pre(gate[2][v]);
        float og = sig_pre(gate[3][v]);
        float cn = fg * c0[v] + ig * gg;
        c0[v] = cn;
        h0w[(mrow + v) * HP + u] = (_Float16)(og * tanh_nat(cn));
    }

    // split barrier: arrive, then hide the wait behind static weight loads
    wg_barrier_signal();
    v16h wf1a0 = load_bfrag_lds(sw1p, u, lh * 16);
    v16h wf1b0 = load_bfrag_lds(sw1p, u, 32 + lh * 16);
    v16h wf2a0 = load_bfrag_lds(sw2p, u, lh * 16);
    v16h wf2b0 = load_bfrag_lds(sw2p, u, 32 + lh * 16);
    wg_barrier_wait();

    // ---------- layer 1: gates = h0n @ Wih1^T + h1 @ Whh1^T + b1 ----------
    v16h a2[2];
#pragma unroll
    for (int ks = 0; ks < 2; ++ks)
        a2[ks] = load_afrag(h0w, arow, ks * 32 + lh * 16);

#pragma unroll
    for (int q = 0; q < 4; ++q) {
        const int g = q * Hh + u;
        v16h f1a, f1b, f2a, f2b;
        if (q == 0) { f1a = wf1a0; f1b = wf1b0; f2a = wf2a0; f2b = wf2b0; }
        else {
            f1a = load_bfrag_lds(sw1p, g, lh * 16);
            f1b = load_bfrag_lds(sw1p, g, 32 + lh * 16);
            f2a = load_bfrag_lds(sw2p, g, lh * 16);
            f2b = load_bfrag_lds(sw2p, g, 32 + lh * 16);
        }
        v8f c;
#pragma unroll
        for (int v = 0; v < 8; ++v) c[v] = bias1[q];
        c = __builtin_amdgcn_wmma_f32_16x16x32_f16(false, a2[0], false, f1a, (short)0, c, false, false);
        c = __builtin_amdgcn_wmma_f32_16x16x32_f16(false, a2[1], false, f1b, (short)0, c, false, false);
        c = __builtin_amdgcn_wmma_f32_16x16x32_f16(false, a1[0], false, f2a, (short)0, c, false, false);
        c = __builtin_amdgcn_wmma_f32_16x16x32_f16(false, a1[1], false, f2b, (short)0, c, false, false);
        gate[q] = c;
    }
#pragma unroll
    for (int v = 0; v < 8; ++v) {
        float ig = sig_pre(gate[0][v]);
        float fg = sig_pre(gate[1][v]);
        float gg = tanh_pre(gate[2][v]);
        float og = sig_pre(gate[3][v]);
        float cn = fg * c1[v] + ig * gg;
        c1[v] = cn;
        float hn = og * tanh_nat(cn);
        h1w[(mrow + v) * HP + u] = (_Float16)hn;
        acc[v] += hn * wfct;          // fc tap; only u==63 lanes' acc is used
    }

    // make new h0/h1 visible for the next timestep
    wg_barrier_signal();
    wg_barrier_wait();
}

__global__ __launch_bounds__(256) void lstm2_wmma_kernel(
    const float* __restrict__ x,
    const float* __restrict__ w_ih0,
    const float* __restrict__ w_hh0,
    const float* __restrict__ b_ih0,
    const float* __restrict__ b_hh0,
    const float* __restrict__ w_ih1,
    const float* __restrict__ w_hh1,
    const float* __restrict__ b_ih1,
    const float* __restrict__ b_hh1,
    const float* __restrict__ w_fc,
    const float* __restrict__ b_fc,
    float* __restrict__ out)
{
    const int n    = blockIdx.x;      // channel
    const int tid  = threadIdx.x;
    const int wv   = tid >> 5;        // wave 0..7 (wave32)
    const int lane = tid & 31;
    const int lh   = lane >> 4;       // lane half (0/1)
    const int l15  = lane & 15;
    const int mt   = wv >> 2;         // batch-row tile 0..1
    const int qcol = wv & 3;          // u-block; this wave owns all 4 gates for it
    const int u    = qcol * 16 + l15; // hidden unit column 0..63

    // double-buffered hidden state (ping-pong; phase is compile-time at call sites)
    __shared__ __align__(16) _Float16 sh0[2][Bm][HP];  // layer-0 hidden (f16)
    __shared__ __align__(16) _Float16 sh1[2][Bm][HP];  // layer-1 hidden (f16)
    __shared__ __align__(16) _Float16 sw1[Gg][HP];     // W_ih1 f16 (pre-scaled)
    __shared__ __align__(16) _Float16 sw2[Gg][HP];     // W_hh1 f16 (pre-scaled)
    __shared__ __align__(16) float    sxT[64][Bm];     // x chunk, [t][b]
    __shared__            float       swfc[64];        // w_fc chunk

    const float* Whh0 = w_hh0 + (size_t)n * Gg * Hh;
    const float* Wih1 = w_ih1 + (size_t)n * Gg * Hh;
    const float* Whh1 = w_hh1 + (size_t)n * Gg * Hh;

    // zero-init both phases of hidden state
    for (int i = tid; i < 2 * Bm * HP; i += 256) {
        (&sh0[0][0][0])[i] = (_Float16)0.0f;
        (&sh1[0][0][0])[i] = (_Float16)0.0f;
    }

    // one-time: convert layer-1 weights f32 -> f16 into LDS, pre-scaled by
    // LOG2E (sigmoid gates) or 2*LOG2E (tanh gate g, rows 128..191)
    {
        const int g = tid;                       // 256 threads == 256 rows
        const float scale = ((g >> 6) == 2) ? (2.0f * LOG2E) : LOG2E;
        const float* s1 = Wih1 + g * Hh;
        const float* s2 = Whh1 + g * Hh;
        _Float16* d1 = &sw1[0][0] + g * HP;
        _Float16* d2 = &sw2[0][0] + g * HP;
#pragma unroll
        for (int k = 0; k < Hh; k += 8) {
            v4f a0 = *(const v4f*)(s1 + k);
            v4f a1 = *(const v4f*)(s1 + k + 4);
            v4f b0 = *(const v4f*)(s2 + k);
            v4f b1 = *(const v4f*)(s2 + k + 4);
            v8h ha, hb;
#pragma unroll
            for (int i = 0; i < 4; ++i) {
                ha[i] = (_Float16)(a0[i] * scale); ha[4 + i] = (_Float16)(a1[i] * scale);
                hb[i] = (_Float16)(b0[i] * scale); hb[4 + i] = (_Float16)(b1[i] * scale);
            }
            *(v8h*)(d1 + k) = ha;
            *(v8h*)(d2 + k) = hb;
        }
    }

    // ---- register-resident layer-0 weight fragments (loop-invariant, pre-scaled) ----
    v16h wb0[4][2];                       // [gate][k-step] = 64 VGPRs
    float bias0[4], bias1[4], wxc[4];
#pragma unroll
    for (int q = 0; q < 4; ++q) {
        const int g = q * Hh + u;
        const float qs = (q == 2) ? (2.0f * LOG2E) : LOG2E;
        bias0[q] = (b_ih0[n * Gg + g] + b_hh0[n * Gg + g]) * qs;
        bias1[q] = (b_ih1[n * Gg + g] + b_hh1[n * Gg + g]) * qs;
        wxc[q]   = w_ih0[n * Gg + g] * qs;   // w_ih0 is (N, 4H, 1)
#pragma unroll
        for (int ks = 0; ks < 2; ++ks)
            wb0[q][ks] = load_wfrag_g(Whh0, g, ks * 32 + lh * 16, qs);
    }

    float c0[8], c1[8], acc[8];           // cell state + fc accumulator, in regs
#pragma unroll
    for (int v = 0; v < 8; ++v) { c0[v] = 0.f; c1[v] = 0.f; acc[v] = 0.f; }

    const int mrow = mt * 16 + lh * 8;    // first C-fragment row (batch b)
    const int arow = mt * 16 + l15;       // A-fragment row for this lane
    const int xb   = tid >> 3;            // refill: batch row
    const int xi   = (tid & 7) * 8;       // refill: t offset in chunk
    const float* xrow = x + ((size_t)xb * Nn + n) * Tt;

    __syncthreads();

    for (int tc = 0; tc < Tt; tc += 64) {
        // stage 64 timesteps of x (per b) and w_fc into LDS
        {
            v4f x0 = *(const v4f*)(xrow + tc + xi);
            v4f x1 = *(const v4f*)(xrow + tc + xi + 4);
#pragma unroll
            for (int j = 0; j < 4; ++j) {
                sxT[xi + j][xb]     = x0[j];
                sxT[xi + 4 + j][xb] = x1[j];
            }
            if (tid < 64) swfc[tid] = w_fc[tc + tid];
            if (tc + 64 < Tt) __builtin_prefetch(xrow + tc + 64, 0, 0);
        }
        __syncthreads();

        // unrolled by 2 so the ping-pong phase is a compile-time constant:
        // even step reads phase 0 / writes phase 1; odd step the reverse.
        for (int tt = 0; tt < 64; tt += 2) {
            lstm_step(&sh0[0][0][0], &sh0[1][0][0], &sh1[0][0][0], &sh1[1][0][0],
                      &sw1[0][0], &sw2[0][0], sxT[tt], swfc[tt],
                      wb0, bias0, bias1, wxc, c0, c1, acc, arow, mrow, u, lh);
            lstm_step(&sh0[1][0][0], &sh0[0][0][0], &sh1[1][0][0], &sh1[0][0][0],
                      &sw1[0][0], &sw2[0][0], sxT[tt + 1], swfc[tt + 1],
                      wb0, bias0, bias1, wxc, c0, c1, acc, arow, mrow, u, lh);
        }
    }

    // out[b,n] = sum_t h1[t,b,n,63]*w_fc[t] + b_fc
    if (u == Hh - 1) {
        const float bf = b_fc[0];
#pragma unroll
        for (int v = 0; v < 8; ++v) {
            const int b = mrow + v;
            out[b * Nn + n] = acc[v] + bf;
        }
    }
}

extern "C" void kernel_launch(void* const* d_in, const int* in_sizes, int n_in,
                              void* d_out, int out_size, void* d_ws, size_t ws_size,
                              hipStream_t stream) {
    (void)in_sizes; (void)n_in; (void)d_ws; (void)ws_size; (void)out_size;
    const float* x     = (const float*)d_in[0];
    const float* w_ih0 = (const float*)d_in[1];
    const float* w_hh0 = (const float*)d_in[2];
    const float* b_ih0 = (const float*)d_in[3];
    const float* b_hh0 = (const float*)d_in[4];
    const float* w_ih1 = (const float*)d_in[5];
    const float* w_hh1 = (const float*)d_in[6];
    const float* b_ih1 = (const float*)d_in[7];
    const float* b_hh1 = (const float*)d_in[8];
    const float* w_fc  = (const float*)d_in[9];
    const float* b_fc  = (const float*)d_in[10];
    float* out = (float*)d_out;

    dim3 grid(Nn), block(256);
    hipLaunchKernelGGL(lstm2_wmma_kernel, grid, block, 0, stream,
                       x, w_ih0, w_hh0, b_ih0, b_hh0,
                       w_ih1, w_hh1, b_ih1, b_hh1, w_fc, b_fc, out);
}
